// FlowmatchingActionHead_56951266344967
// MI455X (gfx1250) — matrix operand order repaired and checked by reading, not compiled, not run on hardware
//
#include <hip/hip_runtime.h>
#include <stdint.h>

// ---------------------------------------------------------------------------
// FlowmatchingActionHead action encoder for MI455X (gfx1250, wave32, WMMA).
// fp32 values split into bf16 hi + bf16 lo; products accumulated as
// hi*hi + hi*lo + lo*hi in fp32 WMMA accumulators (~fp32 accuracy, 3x bf16
// FLOPs => ~44 GFLOP total, balanced against the 114 MB / 23.3 TB/s = ~5us
// weight-traffic floor; weights fit entirely in the 192 MB L2).
// ---------------------------------------------------------------------------

#define B_    64
#define T_    16
#define AD    32
#define H_    1536
#define NCAT  4
#define BN    128     // N columns per block
#define LDB   40      // LDS row stride (32 K + 8 pad bf16) => 80B, 16B-aligned

typedef __bf16 bf16_t;
typedef __attribute__((ext_vector_type(16))) __bf16 v16bf;
typedef __attribute__((ext_vector_type(8)))  float  v8f;

union Frag16 { v16bf v; uint4 q[2]; };
union Pack8  { uint4 q; bf16_t e[8]; };

__device__ __forceinline__ void split_bf16(float v, bf16_t& hi, bf16_t& lo) {
  hi = (bf16_t)v;
  lo = (bf16_t)(v - (float)hi);
}

// ---------------------------------------------------------------------------
// Kernel 0: stable sort of batches by category (64 elements, trivial).
// ---------------------------------------------------------------------------
__global__ void perm_kernel(const int* __restrict__ cat_ids,
                            int* __restrict__ perm, int* __restrict__ cats) {
  if (threadIdx.x == 0) {
    int idx = 0;
    for (int c = 0; c < NCAT; ++c)
      for (int b = 0; b < B_; ++b)
        if (cat_ids[b] == c) { perm[idx] = b; cats[idx] = c; ++idx; }
  }
}

// ---------------------------------------------------------------------------
// Kernel 1: layer-1 (K=32, exact fp32 VALU) + sinusoidal PE + concat,
// stored as bf16 hi/lo in category-sorted row order. One block per batch.
// ---------------------------------------------------------------------------
__global__ void __launch_bounds__(256)
embed_kernel(const float* __restrict__ actions, const int* __restrict__ timesteps,
             const float* __restrict__ W1, const float* __restrict__ b1,
             const int* __restrict__ perm, const int* __restrict__ cats,
             bf16_t* __restrict__ xhi, bf16_t* __restrict__ xlo) {
  __shared__ __align__(16) float  act[T_ * AD];
  __shared__ __align__(16) bf16_t tau_hi[H_];
  __shared__ __align__(16) bf16_t tau_lo[H_];

  const int sb  = blockIdx.x;
  const int b   = perm[sb];
  const int c   = cats[sb];
  const int tid = threadIdx.x;

  for (int i = tid; i < T_ * AD; i += blockDim.x)
    act[i] = actions[(size_t)b * T_ * AD + i];

  // sinusoidal PE of the broadcast timestep (identical for all t in batch)
  const float tval = (float)timesteps[b];
  const float lg   = logf(10000.0f);
  for (int h = tid; h < H_; h += blockDim.x) {
    const int   i   = (h < H_ / 2) ? h : h - H_ / 2;
    const float f   = expf(-lg * (float)i / (float)(H_ / 2));
    const float ang = tval * f;
    const float v   = (h < H_ / 2) ? sinf(ang) : cosf(ang);
    split_bf16(v, tau_hi[h], tau_lo[h]);
  }
  __syncthreads();

  const float* W1c = W1 + (size_t)c * AD * H_;
  const float* b1c = b1 + (size_t)c * H_;
  const int K2 = 2 * H_;

  for (int i = tid; i < T_ * H_; i += blockDim.x) {
    const int t = i / H_, h = i % H_;           // consecutive tid -> consecutive h
    float sum = b1c[h];
#pragma unroll
    for (int d = 0; d < AD; ++d) sum += act[t * AD + d] * W1c[d * H_ + h];
    bf16_t hi, lo; split_bf16(sum, hi, lo);
    const size_t row = (size_t)(sb * T_ + t);
    xhi[row * K2 + h] = hi;
    xlo[row * K2 + h] = lo;
    xhi[row * K2 + H_ + h] = tau_hi[h];
    xlo[row * K2 + H_ + h] = tau_lo[h];
  }
}

// ---------------------------------------------------------------------------
// Kernels 2/3: tiled WMMA GEMM.  BM=16 (one sorted batch => one category),
// BN=128, BK=32.  4 waves/block, each wave owns two 16x16 C tiles.
// W tile (32xBN fp32) is double-buffered: global loads for tile t+1 are
// issued before the (single) barrier of tile t, so HBM latency overlaps the
// WMMA phase.  Staging uses fully-unrolled constant-offset global_load_b32
// (offset k*6144 fits the 24-bit immediate) and packed ds_store_b128.
// mode 0: +bias, swish, emit bf16 hi/lo.  mode 1: +bias, fp32 to d_out with
// rows scattered back to original batch order via perm.
// ---------------------------------------------------------------------------
__global__ void __launch_bounds__(128)
gemm_kernel(const bf16_t* __restrict__ Ahi, const bf16_t* __restrict__ Alo, int K,
            const float* __restrict__ W, const float* __restrict__ bias,
            const int* __restrict__ perm, const int* __restrict__ cats, int mode,
            bf16_t* __restrict__ outHi, bf16_t* __restrict__ outLo,
            float* __restrict__ outF) {
  __shared__ __align__(16) bf16_t Bhi[2][BN * LDB];
  __shared__ __align__(16) bf16_t Blo[2][BN * LDB];

  const int sb    = blockIdx.y;              // sorted batch == 16-row M tile
  const int nbase = blockIdx.x * BN;
  const int c     = cats[sb];
  const float* Wc = W + (size_t)c * K * H_;

  const int tid  = threadIdx.x;
  const int wave = tid >> 5;
  const int lane = tid & 31;
  const int l16  = lane & 15;
  const int kh   = lane >> 4;                // which K half this lane holds

  v8f acc0 = {}; v8f acc1 = {};
  const size_t arow = (size_t)(sb * T_ + l16) * (size_t)K;  // A row, this lane
  const bf16_t* paHi = Ahi + arow + kh * 8;
  const bf16_t* paLo = Alo + arow + kh * 8;

  // prologue: tile 0 of W column n = tid into registers (coalesced across tid)
  const float* src = Wc + nbase + tid;
  float w[32];
#pragma unroll
  for (int k = 0; k < 32; ++k) w[k] = src[k * H_];   // constant imm offsets
  const float* snext = src + 32 * (size_t)H_;

  int p = 0;
  for (int kb = 0; kb < K; kb += 32, p ^= 1) {
    // --- pack current tile regs -> LDS buffer p (8 x ds_store_b128)
    {
      bf16_t* dh = &Bhi[p][tid * LDB];
      bf16_t* dl = &Blo[p][tid * LDB];
#pragma unroll
      for (int g = 0; g < 4; ++g) {
        Pack8 ph, pl;
#pragma unroll
        for (int j = 0; j < 8; ++j) split_bf16(w[g * 8 + j], ph.e[j], pl.e[j]);
        *(uint4*)(dh + g * 8) = ph.q;
        *(uint4*)(dl + g * 8) = pl.q;
      }
    }
    // --- issue next tile's global loads; consumed next iteration
    if (kb + 32 < K) {
#pragma unroll
      for (int k = 0; k < 32; ++k) w[k] = snext[k * H_];
      snext += 32 * (size_t)H_;
    }
    __syncthreads();   // buf p staged everywhere; buf p^1 free to overwrite

    // --- A fragments (16-bit A 16x32 layout): lane<16 holds K {0..7,16..23},
    //     lane>=16 holds K {8..15,24..31} of row M=l16.
    Frag16 ahi, alo;
    ahi.q[0] = *(const uint4*)(paHi + kb);
    ahi.q[1] = *(const uint4*)(paHi + kb + 16);
    alo.q[0] = *(const uint4*)(paLo + kb);
    alo.q[1] = *(const uint4*)(paLo + kb + 16);

    // --- B fragments from LDS (lane<16: K {0..7,16..23}, lane>=16: K {8..15,24..31}, N=l16)
#pragma unroll
    for (int s = 0; s < 2; ++s) {
      const int n = wave * 32 + s * 16 + l16;
      Frag16 bhi, blo;
      const bf16_t* pb = &Bhi[p][n * LDB + kh * 8];
      bhi.q[0] = *(const uint4*)(pb);
      bhi.q[1] = *(const uint4*)(pb + 16);
      const bf16_t* pl = &Blo[p][n * LDB + kh * 8];
      blo.q[0] = *(const uint4*)(pl);
      blo.q[1] = *(const uint4*)(pl + 16);

      v8f acc = s ? acc1 : acc0;
      acc = __builtin_amdgcn_wmma_f32_16x16x32_bf16(false, ahi.v, false, bhi.v,
                                                    (short)0, acc, false, false);
      acc = __builtin_amdgcn_wmma_f32_16x16x32_bf16(false, ahi.v, false, blo.v,
                                                    (short)0, acc, false, false);
      acc = __builtin_amdgcn_wmma_f32_16x16x32_bf16(false, alo.v, false, bhi.v,
                                                    (short)0, acc, false, false);
      if (s) acc1 = acc; else acc0 = acc;
    }
  }

  // --- epilogue.  C layout: VGPR r, lane<16 -> M=r, lane>=16 -> M=r+8, N=l16.
  const float* bc = bias + (size_t)c * H_;
  const int obatch = (mode == 1) ? perm[sb] : sb;
#pragma unroll
  for (int s = 0; s < 2; ++s) {
    const v8f acc = s ? acc1 : acc0;
    const int n = nbase + wave * 32 + s * 16 + l16;
    const float bv = bc[n];
#pragma unroll
    for (int r = 0; r < 8; ++r) {
      const int m = r + kh * 8;
      const float v = acc[r] + bv;
      if (mode == 0) {
        const float sw = v / (1.0f + expf(-v));      // swish
        bf16_t hi, lo; split_bf16(sw, hi, lo);
        const size_t off = (size_t)(sb * T_ + m) * H_ + n;
        outHi[off] = hi; outLo[off] = lo;
      } else {
        const size_t off = ((size_t)obatch * T_ + m) * H_ + n;
        outF[off] = v;
      }
    }
  }
}

// ---------------------------------------------------------------------------
extern "C" void kernel_launch(void* const* d_in, const int* in_sizes, int n_in,
                              void* d_out, int out_size, void* d_ws, size_t ws_size,
                              hipStream_t stream) {
  const float* actions   = (const float*)d_in[0];
  const int*   timesteps = (const int*)d_in[1];
  const int*   cat_ids   = (const int*)d_in[2];
  const float* W1        = (const float*)d_in[3];
  const float* b1        = (const float*)d_in[4];
  const float* W2        = (const float*)d_in[5];
  const float* b2        = (const float*)d_in[6];
  const float* W3        = (const float*)d_in[7];
  const float* b3        = (const float*)d_in[8];
  float* out = (float*)d_out;

  // workspace carve-up (~18.9 MB), 256B-aligned segments
  char*   ws   = (char*)d_ws;
  int*    perm = (int*)ws;                       // 64 ints
  int*    cats = (int*)(ws + 256);               // 64 ints
  bf16_t* xhi  = (bf16_t*)(ws + 512);            // 1024 x 3072 bf16
  bf16_t* xlo  = xhi + (size_t)1024 * 3072;
  bf16_t* hhi  = xlo + (size_t)1024 * 3072;      // 1024 x 1536 bf16
  bf16_t* hlo  = hhi + (size_t)1024 * 1536;

  perm_kernel<<<1, 32, 0, stream>>>(cat_ids, perm, cats);
  embed_kernel<<<B_, 256, 0, stream>>>(actions, timesteps, W1, b1, perm, cats,
                                       xhi, xlo);
  // layer 2: (1024 x 3072) @ (3072 x 1536), swish, emit bf16 hi/lo
  gemm_kernel<<<dim3(H_ / BN, B_), 128, 0, stream>>>(xhi, xlo, 2 * H_, W2, b2,
                                                     perm, cats, 0, hhi, hlo,
                                                     nullptr);
  // layer 3: (1024 x 1536) @ (1536 x 1536), fp32 out, unpermuted rows
  gemm_kernel<<<dim3(H_ / BN, B_), 128, 0, stream>>>(hhi, hlo, H_, W3, b3,
                                                     perm, cats, 1, nullptr,
                                                     nullptr, out);
}
